// SelfAttention_38250978738282
// MI455X (gfx1250) — matrix operand order repaired and checked by reading, not compiled, not run on hardware
//
#include <hip/hip_runtime.h>
#include <hip/hip_bf16.h>

typedef __attribute__((ext_vector_type(16))) _Float16 v16h;
typedef __attribute__((ext_vector_type(8)))  _Float16 v8h;
typedef __attribute__((ext_vector_type(8)))  float    v8f;

#define B_  4
#define C_  256
#define C8_ 32
#define HP_ 64
#define WP_ 64
#define N_  4096   // HP_*WP_
#define H_  128
#define W_  128
#define LOG2E_ 1.4426950408889634f

// butterfly (lane ^ m within each 16-lane row) via v_permlane16_b32: pure VALU.
__device__ __forceinline__ float permlane16_f(float v, unsigned s0, unsigned s1) {
    int i = __builtin_bit_cast(int, v);
    int r = __builtin_amdgcn_permlane16(i, i, s0, s1, false, false);
    return __builtin_bit_cast(float, r);
}
__device__ __forceinline__ float rowmax16(float mx) {
    mx = fmaxf(mx, permlane16_f(mx, 0x67452301u, 0xEFCDAB89u)); // xor 1
    mx = fmaxf(mx, permlane16_f(mx, 0x54761032u, 0xDCFE98BAu)); // xor 2
    mx = fmaxf(mx, permlane16_f(mx, 0x32107654u, 0xBA98FEDCu)); // xor 4
    mx = fmaxf(mx, permlane16_f(mx, 0xFEDCBA98u, 0x76543210u)); // xor 8
    return mx;
}

// ---------------------------------------------------------------------------
// Kernel 1: 2x2 avg-pool + 1x1 conv projections (q,k,v) in one pass.
//   Q stored PRE-SCALED by log2(e) so attention softmax can use exp2.
//   Q,K as [B][N][32] f16 (per-position rows -> contiguous frag loads)
//   V  as [B][C][N] f16   (channel-major -> contiguous 32B B-frag loads)
// ---------------------------------------------------------------------------
__global__ __launch_bounds__(256) void qkv_kernel(
    const float* __restrict__ x,
    const float* __restrict__ wq, const float* __restrict__ bq,
    const float* __restrict__ wk, const float* __restrict__ bk,
    const float* __restrict__ wv, const float* __restrict__ bv,
    _Float16* __restrict__ Qb, _Float16* __restrict__ Kb,
    _Float16* __restrict__ Vt)
{
    __shared__ float xs[C_][17];          // padded: stride 17 vs 64 banks
    const int blk = blockIdx.x;           // 0 .. B_*(N_/16)-1
    const int b   = blk >> 8;
    const int n0  = (blk & 255) << 4;     // 16 consecutive positions, one pooled row
    const int hp  = n0 >> 6;
    const int wp0 = n0 & 63;
    const int c   = threadIdx.x;

    const float* xb = x + (((size_t)b * C_ + c) * H_) * W_;
    #pragma unroll 4
    for (int p = 0; p < 16; ++p) {
        const float* px = xb + (size_t)(2 * hp) * W_ + 2 * (wp0 + p);
        xs[c][p] = 0.25f * (px[0] + px[1] + px[W_] + px[W_ + 1]);
    }
    __syncthreads();

    // V projection: thread c -> output channel c, 16 positions
    float acc[16];
    const float bvc = bv[c];
    #pragma unroll
    for (int p = 0; p < 16; ++p) acc[p] = bvc;
    const float* wr = wv + (size_t)c * C_;
    for (int i = 0; i < C_; ++i) {
        const float w = wr[i];
        #pragma unroll
        for (int p = 0; p < 16; ++p) acc[p] = fmaf(w, xs[i][p], acc[p]);
    }
    _Float16* vdst = Vt + ((size_t)b * C_ + c) * N_ + n0;
    #pragma unroll
    for (int p = 0; p < 16; ++p) vdst[p] = (_Float16)acc[p];

    // Q/K projections on threads 0..63 (Q gets the log2e fold)
    if (c < 64) {
        const int  o    = c & 31;
        const bool isQ  = (c < 32);
        const float* w2 = (isQ ? wq : wk) + (size_t)o * C_;
        float a2[16];
        const float bb = isQ ? bq[o] : bk[o];
        #pragma unroll
        for (int p = 0; p < 16; ++p) a2[p] = bb;
        for (int i = 0; i < C_; ++i) {
            const float w = w2[i];
            #pragma unroll
            for (int p = 0; p < 16; ++p) a2[p] = fmaf(w, xs[i][p], a2[p]);
        }
        const float scale = isQ ? LOG2E_ : 1.0f;
        _Float16* dst = (isQ ? Qb : Kb) + ((size_t)b * N_ + n0) * C8_ + o;
        #pragma unroll
        for (int p = 0; p < 16; ++p) dst[(size_t)p * C8_] = (_Float16)(a2[p] * scale);
    }
}

// ---------------------------------------------------------------------------
// Kernel 2: flash attention, 19 x v_wmma_f32_16x16x32_f16 per 32-key step:
//   2 for S = Q*K^T (base-2 logits), 16 for O += P*V, 1 for the softmax
//   denominator via a constant all-ones B fragment (l = P x 1-vector).
// Row-max via v_permlane16 butterflies; accumulator rescale skipped on the
// (common) steps where no row max changes, guarded by wave-uniform __any.
// ---------------------------------------------------------------------------
__global__ __launch_bounds__(128) void attn_kernel(
    const _Float16* __restrict__ Qb, const _Float16* __restrict__ Kb,
    const _Float16* __restrict__ Vt, float* __restrict__ Oattn)
{
    __shared__ _Float16 plds[4][16][32];   // per-wave P staging (C-layout -> A-layout)
    const int wave = threadIdx.x >> 5;
    const int lane = threadIdx.x & 31;
    const int half = lane >> 4;            // which 16-lane half
    const int lm   = lane & 15;

    const int tile = blockIdx.x * 4 + wave;   // 0 .. B_*N_/16 - 1
    const int b    = tile >> 8;
    const int q0   = (tile & 255) << 4;

    // Q A-fragment: lane holds row q0+lm, K-dims half*16+0..15
    const _Float16* qp = Qb + ((size_t)b * N_ + q0 + lm) * C8_;
    const v8h qlo = *(const v8h*)(qp + half * 16);
    const v8h qhi = *(const v8h*)(qp + half * 16 + 8);
    v16h aq;
    #pragma unroll
    for (int i = 0; i < 8; ++i) { aq[i] = qlo[i]; aq[8 + i] = qhi[i]; }

    // constant all-ones B fragment: denominator = P x ones (every col identical)
    v16h bones;
    #pragma unroll
    for (int i = 0; i < 16; ++i) bones[i] = (_Float16)1.0f;

    v8f oacc[16];                           // 16x256 f32 O accumulator
    v8f oden = {};                          // softmax denominator accumulator
    #pragma unroll
    for (int t = 0; t < 16; ++t)
        #pragma unroll
        for (int r = 0; r < 8; ++r) oacc[t][r] = 0.0f;
    #pragma unroll
    for (int r = 0; r < 8; ++r) oden[r] = 0.0f;

    float rm[8];                            // running max per row (row = 8*half + r)
    #pragma unroll
    for (int r = 0; r < 8; ++r) rm[r] = -1e30f;

    const _Float16* kbase = Kb + (size_t)b * N_ * C8_;
    const _Float16* vbase = Vt + (size_t)b * C_ * N_;

    for (int j0 = 0; j0 < N_; j0 += 32) {
        // K B-fragments for key tiles [j0, j0+16) and [j0+16, j0+32)
        const v16h bk0 = *(const v16h*)(kbase + (size_t)(j0 + lm)      * C8_ + half * 16);
        const v16h bk1 = *(const v16h*)(kbase + (size_t)(j0 + 16 + lm) * C8_ + half * 16);

        v8f s0 = {}; v8f s1 = {};
        s0 = __builtin_amdgcn_wmma_f32_16x16x32_f16(false, aq, false, bk0, (short)0, s0, false, false);
        s1 = __builtin_amdgcn_wmma_f32_16x16x32_f16(false, aq, false, bk1, (short)0, s1, false, false);

        // online softmax per row (C-layout: VGPR r == row 8*half+r; keys across lanes)
        float alpha[8];
        bool  upd = false;
        #pragma unroll
        for (int r = 0; r < 8; ++r) {
            const float a  = s0[r];
            const float bb = s1[r];
            const float mx   = rowmax16(fmaxf(a, bb));     // VALU butterfly
            const float mnew = fmaxf(rm[r], mx);
            upd      = upd || (mnew > rm[r]);
            alpha[r] = exp2f(rm[r] - mnew);                // single v_exp_f32
            rm[r]    = mnew;
            const float p0 = exp2f(a  - mnew);
            const float p1 = exp2f(bb - mnew);
            plds[wave][half * 8 + r][lm]      = (_Float16)p0;
            plds[wave][half * 8 + r][16 + lm] = (_Float16)p1;
        }

        // rescale running accumulators only when some row max moved (exact:
        // alpha==1 for all rows otherwise). Wave-uniform branch, no WMMA inside.
        if (__any(upd)) {
            #pragma unroll
            for (int t = 0; t < 16; ++t)
                #pragma unroll
                for (int r = 0; r < 8; ++r) oacc[t][r] *= alpha[r];
            #pragma unroll
            for (int r = 0; r < 8; ++r) oden[r] *= alpha[r];
        }

        // reload P as A-fragment (same addressing pattern as Q)
        const v8h plo = *(const v8h*)&plds[wave][lm][half * 16];
        const v8h phi = *(const v8h*)&plds[wave][lm][half * 16 + 8];
        v16h ap;
        #pragma unroll
        for (int i = 0; i < 8; ++i) { ap[i] = plo[i]; ap[8 + i] = phi[i]; }

        // denominator: l += P x ones
        oden = __builtin_amdgcn_wmma_f32_16x16x32_f16(false, ap, false, bones, (short)0, oden, false, false);

        // O(16x256) += P(16x32) x V(32x256): 16 column tiles
        #pragma unroll
        for (int t = 0; t < 16; ++t) {
            const v16h bvf = *(const v16h*)(vbase + ((size_t)(t * 16 + lm)) * N_ + j0 + half * 16);
            oacc[t] = __builtin_amdgcn_wmma_f32_16x16x32_f16(false, ap, false, bvf, (short)0, oacc[t], false, false);
        }
    }

    // epilogue: divide by denominator (already in matching C-layout), write [B][C][N]
    float inv[8];
    #pragma unroll
    for (int r = 0; r < 8; ++r) inv[r] = 1.0f / oden[r];
    #pragma unroll
    for (int t = 0; t < 16; ++t)
        #pragma unroll
        for (int r = 0; r < 8; ++r)
            Oattn[((size_t)b * C_ + t * 16 + lm) * N_ + q0 + half * 8 + r] = oacc[t][r] * inv[r];
}

// ---------------------------------------------------------------------------
// Kernel 3: bilinear x2 upsample (half-pixel, edge-clamped) + gamma*up + x
// ---------------------------------------------------------------------------
__global__ __launch_bounds__(256) void upsample_kernel(
    const float* __restrict__ O, const float* __restrict__ x,
    const float* __restrict__ gamma, float* __restrict__ out, int total)
{
    const int idx = blockIdx.x * blockDim.x + threadIdx.x;
    if (idx >= total) return;
    const int wq = idx & (W_ - 1);
    const int hq = (idx >> 7) & (H_ - 1);
    const int bc = idx >> 14;               // b*C_ + c

    const float sy = (hq + 0.5f) * 0.5f - 0.5f;
    const float sx = (wq + 0.5f) * 0.5f - 0.5f;
    const float fy = floorf(sy), fx = floorf(sx);
    const float wy = sy - fy,    wx = sx - fx;
    int y0 = (int)fy, x0 = (int)fx;
    const int y1 = min(y0 + 1, HP_ - 1);
    const int x1 = min(x0 + 1, WP_ - 1);
    y0 = max(y0, 0); x0 = max(x0, 0);

    const float* Ob = O + (size_t)bc * N_;
    const float v00 = Ob[y0 * WP_ + x0];
    const float v01 = Ob[y0 * WP_ + x1];
    const float v10 = Ob[y1 * WP_ + x0];
    const float v11 = Ob[y1 * WP_ + x1];
    const float top = v00 + wx * (v01 - v00);
    const float bot = v10 + wx * (v11 - v10);
    const float val = top + wy * (bot - top);
    out[idx] = gamma[0] * val + x[idx];
}

// ---------------------------------------------------------------------------
extern "C" void kernel_launch(void* const* d_in, const int* in_sizes, int n_in,
                              void* d_out, int out_size, void* d_ws, size_t ws_size,
                              hipStream_t stream)
{
    const float* x     = (const float*)d_in[0];
    const float* wq    = (const float*)d_in[1];
    const float* bq    = (const float*)d_in[2];
    const float* wk    = (const float*)d_in[3];
    const float* bk    = (const float*)d_in[4];
    const float* wv    = (const float*)d_in[5];
    const float* bv    = (const float*)d_in[6];
    const float* gamma = (const float*)d_in[7];
    float* out = (float*)d_out;

    // workspace layout:
    //   Qb [B][N][32] f16 : 1 MB    @ 0
    //   Kb [B][N][32] f16 : 1 MB    @ 1 MB
    //   Vt [B][C][N]  f16 : 8 MB    @ 2 MB
    //   Oattn [B][C][N] f32: 16 MB  @ 10 MB
    char* ws = (char*)d_ws;
    _Float16* Qb    = (_Float16*)(ws);
    _Float16* Kb    = (_Float16*)(ws + (size_t)(1u << 20));
    _Float16* Vt    = (_Float16*)(ws + (size_t)(2u << 20));
    float*    Oattn = (float*)   (ws + (size_t)(10u << 20));

    qkv_kernel<<<B_ * (N_ / 16), 256, 0, stream>>>(x, wq, bq, wk, bk, wv, bv, Qb, Kb, Vt);
    attn_kernel<<<B_ * (N_ / 16) / 4, 128, 0, stream>>>(Qb, Kb, Vt, Oattn);
    const int total = B_ * C_ * H_ * W_;
    upsample_kernel<<<(total + 255) / 256, 256, 0, stream>>>(Oattn, x, gamma, out, total);
}
